// GroupConv1d_37744172597304
// MI455X (gfx1250) — matrix operand order repaired and checked by reading, not compiled, not run on hardware
//
#include <hip/hip_runtime.h>

typedef __bf16 bf16;
typedef __attribute__((ext_vector_type(16))) __bf16 v16bf;
typedef __attribute__((ext_vector_type(8)))  __bf16 v8bf;
typedef __attribute__((ext_vector_type(4)))  __bf16 v4bf;
typedef __attribute__((ext_vector_type(8)))  float  v8f;
typedef __attribute__((ext_vector_type(4)))  float  v4f;

#define C_IN   256
#define C_OUT  256
#define HW     784          // 28*28
#define CHW    (C_IN * HW)
#define NSEG   8
#define LDSW   264          // bf16 elems per weight row (256 + 8 pad) -> 528B = 132 dwords
#define TILES_PER_FRAME 49  // 784 / 16
#define N_TILES 6272        // 128 frames * 49

// out^T tiling: D = A(16oc x 32c) * B(32c x 16px) + C, bf16 in / f32 acc.
// A from LDS-staged bf16 weights (row-major, padded stride, double-buffered),
// B gathered from global x with the temporal-shift frame select applied per
// 32-channel chunk and software-pipelined one chunk ahead.
__global__ __launch_bounds__(256) void tsm_pointwise_gemm(
    const float* __restrict__ x,      // [128, 256, 28, 28]
    const float* __restrict__ w,      // [256, 256]  (out_c, in_c)
    float* __restrict__ out)          // [128, 256, 28, 28]
{
    extern __shared__ bf16 wlds[];    // C_OUT * LDSW bf16 = 135168 B

    const int tid = threadIdx.x;

    // ---- stage weights f32 -> bf16 into LDS (coalesced b128 global loads) ----
    for (int i = tid; i < (C_OUT * C_IN) / 4; i += 256) {
        const int idx = i * 4;
        const int row = idx >> 8;
        const int col = idx & 255;
        v4f v = *(const v4f*)(w + idx);
        v4bf p;
        p[0] = (bf16)v.x; p[1] = (bf16)v.y; p[2] = (bf16)v.z; p[3] = (bf16)v.w;
        *(v4bf*)(&wlds[row * LDSW + col]) = p;   // 8B-aligned ds store
    }
    __syncthreads();

    const int wave = tid >> 5;
    const int lane = tid & 31;
    const int tile = blockIdx.x * 8 + wave;      // grid sized so tile < N_TILES
    const int n    = tile / TILES_PER_FRAME;     // frame index
    const int hw0  = (tile % TILES_PER_FRAME) * 16;
    const int t    = n & (NSEG - 1);             // position within 8-frame clip

    const int lm = lane & 15;                    // row-within-tile selector
    const int lh = lane >> 4;                    // half-wave selector

    // A-fragment per-lane base: oc row = nt*16 + lm; low lanes take K
    // {0..15}, high lanes {16..31} -> base element offset lh*16.
    const bf16* aBase = &wlds[lm * LDSW + lh * 16];

    // B gather: pixel p = hw0 + lm, channels c0 + lh*16 + k (k = 0..15).
    const float* xn = x + (size_t)n * CHW;

    v8f acc[16];
    #pragma unroll
    for (int i = 0; i < 16; ++i) {
        v8f z = {0.f, 0.f, 0.f, 0.f, 0.f, 0.f, 0.f, 0.f};
        acc[i] = z;
    }

    // Gather one 32-channel chunk of shifted x (16 raw f32 per lane).
    // Temporal shift is uniform per chunk (fold == 32 == chunk size).
    auto gatherB = [&](int kc, float f[16]) {
        const float* src = xn;
        bool zero = false;
        if (kc == 0)      { src = xn - CHW; zero = (t == 0); }        // prev frame
        else if (kc == 1) { src = xn + CHW; zero = (t == NSEG - 1); } // next frame
        if (zero) {
            #pragma unroll
            for (int k = 0; k < 16; ++k) f[k] = 0.f;
        } else {
            const float* p = src + (size_t)(kc * 32 + lh * 16) * HW + hw0 + lm;
            #pragma unroll
            for (int k = 0; k < 16; ++k) f[k] = p[(size_t)k * HW]; // 64B/half-wave runs
        }
    };

    float fcur[16];
    gatherB(0, fcur);

    for (int kc = 0; kc < 8; ++kc) {
        // Convert current chunk to the bf16 B fragment.
        v16bf bfrag;
        #pragma unroll
        for (int k = 0; k < 16; ++k) bfrag[k] = (bf16)fcur[k];

        // Prefetch next chunk's raw floats; consumed next iteration, so the
        // global loads overlap the 16 WMMAs + LDS traffic below.
        if (kc < 7) gatherB(kc + 1, fcur);

        const bf16* ap = aBase + kc * 32;
        auto loadA = [&](int nt) -> v16bf {
            const bf16* q = ap + nt * (16 * LDSW);
            v8bf lo = *(const v8bf*)(q);        // K 0..7 of half (16B aligned)
            v8bf hi = *(const v8bf*)(q + 8);    // K 8..15 of half (+16B)
            return __builtin_shufflevector(
                lo, hi, 0,1,2,3,4,5,6,7,8,9,10,11,12,13,14,15);
        };

        // Double-buffer A fragments: issue loads for nt+1 before WMMA nt so
        // the backend can wait with dscnt<=2 instead of a full drain.
        v16bf a_cur = loadA(0);
        #pragma unroll
        for (int nt = 0; nt < 15; ++nt) {
            v16bf a_nxt = loadA(nt + 1);
            acc[nt] = __builtin_amdgcn_wmma_f32_16x16x32_bf16(
                false, a_cur, false, bfrag, (short)0, acc[nt], false, false);
            a_cur = a_nxt;
        }
        acc[15] = __builtin_amdgcn_wmma_f32_16x16x32_bf16(
            false, a_cur, false, bfrag, (short)0, acc[15], false, false);
    }

    // D layout: lane -> N (= pixel), VGPR r -> M row r (low lanes) / r+8 (high).
    // Stores are 64B-contiguous per half-wave in NCHW out.
    float* obase = out + (size_t)n * (C_OUT * HW) + hw0 + lm;
    #pragma unroll
    for (int nt = 0; nt < 16; ++nt) {
        #pragma unroll
        for (int r = 0; r < 8; ++r) {
            const int oc = nt * 16 + r + 8 * lh;
            obase[(size_t)oc * HW] = acc[nt][r];
        }
    }
}

extern "C" void kernel_launch(void* const* d_in, const int* in_sizes, int n_in,
                              void* d_out, int out_size, void* d_ws, size_t ws_size,
                              hipStream_t stream) {
    const float* x = (const float*)d_in[0];   // [128,256,28,28] f32
    const float* w = (const float*)d_in[1];   // [256,256] f32
    float* out = (float*)d_out;               // [128,256,28,28] f32

    const int blocks = N_TILES / 8;           // 784 blocks * 8 waves = 6272 tiles
    const size_t lds_bytes = (size_t)C_OUT * LDSW * sizeof(bf16); // 135168
    tsm_pointwise_gemm<<<blocks, 256, lds_bytes, stream>>>(x, w, out);
}
